// AttentionBlock_61435212202712
// MI455X (gfx1250) — matrix operand order repaired and checked by reading, not compiled, not run on hardware
//
#include <hip/hip_runtime.h>

// ---------------------------------------------------------------------------
// MI455X (gfx1250) fused AttentionBlock, bf16 WMMA path.
// B=2, C=512, L=4096, 32 groups, 8 heads, ch=64.
// ---------------------------------------------------------------------------

typedef __bf16 bf16;
typedef __attribute__((ext_vector_type(16))) __bf16 v16bf;
typedef __attribute__((ext_vector_type(8)))  __bf16 v8bf;
typedef __attribute__((ext_vector_type(8)))  float  v8f;

#define CDIM   512
#define LDIM   4096
#define NHEAD  8
#define CH     64
#define OQKV   1536

__device__ __forceinline__ v16bf cat8(v8bf lo, v8bf hi) {
  return __builtin_shufflevector(lo, hi, 0,1,2,3,4,5,6,7,8,9,10,11,12,13,14,15);
}

// A operand 16x32 (M x K), per ISA: lane l -> row M=l%16; h=l/16 selects
// K chunks [8h,8h+8) and [16+8h, 24+8h).
__device__ __forceinline__ v16bf loadA(const bf16* row, int h) {
  return cat8(*(const v8bf*)(row + 8 * h),
              *(const v8bf*)(row + 16 + 8 * h));
}
// B operand 32x16 (K x N): lane l -> col N=l%16; h=l/16 selects
// K chunks [8h,8h+8) and [16+8h, 24+8h) (same order as A).
__device__ __forceinline__ v16bf loadB(const bf16* row, int h) {
  return cat8(*(const v8bf*)(row + 8 * h),
              *(const v8bf*)(row + 16 + 8 * h));
}

__device__ __forceinline__ v8f wmma_bf16(v16bf a, v16bf b, v8f c) {
  return __builtin_amdgcn_wmma_f32_16x16x32_bf16(false, a, false, b,
                                                 (short)0, c, false, false);
}

// ---------------------------------------------------------------------------
// Kernel 0: convert fp32 weights to bf16 (one-shot, L2-resident afterwards)
// ---------------------------------------------------------------------------
__global__ __launch_bounds__(256) void cvt_kernel(const float* __restrict__ wq,
                                                  const float* __restrict__ wp,
                                                  bf16* __restrict__ wq_bf,
                                                  bf16* __restrict__ wp_bf) {
  int i = blockIdx.x * 256 + threadIdx.x;
  if (i < OQKV * CDIM) wq_bf[i] = (bf16)wq[i];
  if (i < CDIM * CDIM) wp_bf[i] = (bf16)wp[i];
}

// ---------------------------------------------------------------------------
// Kernel 1: GroupNorm. One block per (batch, group); writes h transposed
// h_t[b][l][c] in bf16 (row = contraction dim for the QKV GEMM B operand).
// ---------------------------------------------------------------------------
__global__ __launch_bounds__(256) void gn_kernel(const float* __restrict__ x,
                                                 const float* __restrict__ w,
                                                 const float* __restrict__ b,
                                                 bf16* __restrict__ h_t) {
  const int blk = blockIdx.x;
  const int bb = blk >> 5;        // batch
  const int g  = blk & 31;        // group
  const float* xg = x + ((size_t)bb * CDIM + (size_t)g * 16) * LDIM;

  float s = 0.f, ss = 0.f;
  for (int i = threadIdx.x; i < 16 * LDIM; i += 256) {
    float v = xg[i];
    s += v; ss += v * v;
  }
  __shared__ float red0[256];
  __shared__ float red1[256];
  red0[threadIdx.x] = s; red1[threadIdx.x] = ss;
  __syncthreads();
  for (int off = 128; off > 0; off >>= 1) {
    if (threadIdx.x < off) {
      red0[threadIdx.x] += red0[threadIdx.x + off];
      red1[threadIdx.x] += red1[threadIdx.x + off];
    }
    __syncthreads();
  }
  const float mean = red0[0] * (1.f / 65536.f);
  const float var  = red1[0] * (1.f / 65536.f) - mean * mean;
  const float inv  = rsqrtf(var + 1e-5f);

  for (int i = threadIdx.x; i < 16 * LDIM; i += 256) {
    int c = i >> 12;            // 0..15 within group
    int l = i & (LDIM - 1);
    int cg = g * 16 + c;
    float v = (xg[i] - mean) * inv * w[cg] + b[cg];
    h_t[((size_t)bb * LDIM + l) * CDIM + cg] = (bf16)v;
  }
}

// ---------------------------------------------------------------------------
// Kernel 2: QKV GEMM (1536x512 @ 512x4096) + bias, scatter to q_t/k_t/v_n.
// q/k are pre-scaled by 1/ch^0.25 (q additionally by log2(e) so the flash
// kernel can use native base-2 exp). q_t,k_t stored [bh][l][64]; v [bh][64][L].
// One wave per 16x16 output tile, K-loop of 16 bf16 WMMAs.
// ---------------------------------------------------------------------------
__global__ __launch_bounds__(256) void qkv_kernel(const bf16* __restrict__ wq_bf,
                                                  const float* __restrict__ qkv_b,
                                                  const bf16* __restrict__ h_t,
                                                  bf16* __restrict__ q_t,
                                                  bf16* __restrict__ k_t,
                                                  bf16* __restrict__ v_n) {
  const int wave = threadIdx.x >> 5, lane = threadIdx.x & 31;
  const int m = lane & 15, hh = lane >> 4;
  const int ot = blockIdx.x * 8 + wave;  // 0..95
  const int lt = blockIdx.y;             // 0..255
  const int bb = blockIdx.z;             // 0..1
  const int o0 = ot * 16, l0 = lt * 16;

  const bf16* arow = wq_bf + (size_t)(o0 + m) * CDIM;
  const bf16* brow = h_t + ((size_t)bb * LDIM + l0 + m) * CDIM;

  v8f acc = {};
#pragma unroll 4
  for (int c0 = 0; c0 < CDIM; c0 += 32)
    acc = wmma_bf16(loadA(arow + c0, hh), loadB(brow + c0, hh), acc);

  const int head  = o0 / (3 * CH);
  const int rbase = o0 % (3 * CH);
  const int bh = bb * NHEAD + head;
  const int l  = l0 + m;
  const float kscale = 0.3535533906f;                 // ch^-0.25
  const float qscale = kscale * 1.44269504088896f;    // fold log2(e)

#pragma unroll
  for (int r = 0; r < 8; r++) {
    int o  = o0 + r + 8 * hh;
    int rr = rbase + r + 8 * hh;
    float val = acc[r] + qkv_b[o];
    if (rr < CH) {
      q_t[((size_t)bh * LDIM + l) * CH + rr] = (bf16)(val * qscale);
    } else if (rr < 2 * CH) {
      k_t[((size_t)bh * LDIM + l) * CH + (rr - CH)] = (bf16)(val * kscale);
    } else {
      v_n[((size_t)bh * CH + (rr - 2 * CH)) * LDIM + l] = (bf16)val;
    }
  }
}

// ---------------------------------------------------------------------------
// Kernel 3: flash attention per head. 8 waves/WG, each wave owns a 16-row
// t-tile; WG covers 128 t-rows of one head. s is processed in chunks of 64,
// K/V chunks staged cooperatively in padded LDS. Online softmax in base-2.
// ---------------------------------------------------------------------------
__global__ __launch_bounds__(256) void attn_kernel(const bf16* __restrict__ q_t,
                                                   const bf16* __restrict__ k_t,
                                                   const bf16* __restrict__ v_n,
                                                   bf16* __restrict__ a_t) {
  __shared__ bf16 k_sh[64][72];       // [s][c], +8 pad keeps banks spread
  __shared__ bf16 v_sh[64][72];       // [c][s]
  __shared__ bf16 p_sh[8][16][40];    // per-wave D->A relayout scratch

  const int tid = threadIdx.x;
  const int wave = tid >> 5, lane = tid & 31;
  const int m = lane & 15, hh = lane >> 4;
  const int bh = blockIdx.y;                      // 0..15
  const int t0 = blockIdx.x * 128 + wave * 16;

  // Q fragments for this wave's 16 t-rows (A operand, K=64 -> two frags).
  const bf16* qbase = q_t + ((size_t)bh * LDIM + t0 + m) * CH;
  const v16bf aq0 = loadA(qbase, hh);        // c 0..31
  const v16bf aq1 = loadA(qbase + 32, hh);   // c 32..63

  v8f O0 = {}, O1 = {}, O2 = {}, O3 = {};
  float mrow[8], lrow[8];
#pragma unroll
  for (int r = 0; r < 8; r++) { mrow[r] = -1e30f; lrow[r] = 0.f; }

  const bf16* kg = k_t + (size_t)bh * LDIM * CH;
  const bf16* vg = v_n + (size_t)bh * CH * LDIM;

  for (int s0 = 0; s0 < LDIM; s0 += 64) {
    __syncthreads();  // protect LDS reuse from previous iteration's readers
#pragma unroll
    for (int part = 0; part < 2; part++) {            // 512 x 16B chunks each
      int chunk = tid + part * 256;
      int row = chunk >> 3, cc = (chunk & 7) * 8;
      *(v8bf*)&k_sh[row][cc] = *(const v8bf*)(kg + (size_t)(s0 + row) * CH + cc);
      *(v8bf*)&v_sh[row][cc] = *(const v8bf*)(vg + (size_t)row * LDIM + s0 + cc);
    }
    __syncthreads();

#pragma unroll
    for (int pair = 0; pair < 2; pair++) {
      const int sbase = pair * 32;
      // --- S = Q^T K for two adjacent 16-col subtiles (32 s-columns) ---
      v8f Sa = {}, Sb = {};
      {
        const bf16* kr = &k_sh[sbase + m][0];
        Sa = wmma_bf16(aq0, loadB(kr, hh), Sa);
        Sa = wmma_bf16(aq1, loadB(kr + 32, hh), Sa);
        const bf16* kr2 = &k_sh[sbase + 16 + m][0];
        Sb = wmma_bf16(aq0, loadB(kr2, hh), Sb);
        Sb = wmma_bf16(aq1, loadB(kr2 + 32, hh), Sb);
      }
      // --- online softmax (base-2; log2e folded into q) ---
      float alpha[8];
#pragma unroll
      for (int r = 0; r < 8; r++) {
        float mx = fmaxf(Sa[r], Sb[r]);
#pragma unroll
        for (int off = 1; off < 16; off <<= 1)
          mx = fmaxf(mx, __shfl_xor(mx, off, 32));
        float mnew = fmaxf(mrow[r], mx);
        float al = exp2f(mrow[r] - mnew);
        float pa = exp2f(Sa[r] - mnew);
        float pb = exp2f(Sb[r] - mnew);
        float rs = pa + pb;
#pragma unroll
        for (int off = 1; off < 16; off <<= 1)
          rs += __shfl_xor(rs, off, 32);
        lrow[r] = lrow[r] * al + rs;
        mrow[r] = mnew;
        alpha[r] = al;
        Sa[r] = pa; Sb[r] = pb;
      }
#pragma unroll
      for (int r = 0; r < 8; r++) {
        O0[r] *= alpha[r]; O1[r] *= alpha[r];
        O2[r] *= alpha[r]; O3[r] *= alpha[r];
      }
      // --- stage P: D layout (row=vgpr+8h, col=lane) -> A layout via LDS ---
#pragma unroll
      for (int r = 0; r < 8; r++) {
        p_sh[wave][r + 8 * hh][m]      = (bf16)Sa[r];
        p_sh[wave][r + 8 * hh][m + 16] = (bf16)Sb[r];
      }
      asm volatile("s_wait_dscnt 0" ::: "memory");
      const v16bf pfrag = cat8(*(const v8bf*)&p_sh[wave][m][8 * hh],
                               *(const v8bf*)&p_sh[wave][m][16 + 8 * hh]);
      // --- O += P (t x 32s)  x  V^T (32s x 64c), 4 c-subtiles ---
      O0 = wmma_bf16(pfrag, loadB(&v_sh[ 0 + m][sbase], hh), O0);
      O1 = wmma_bf16(pfrag, loadB(&v_sh[16 + m][sbase], hh), O1);
      O2 = wmma_bf16(pfrag, loadB(&v_sh[32 + m][sbase], hh), O2);
      O3 = wmma_bf16(pfrag, loadB(&v_sh[48 + m][sbase], hh), O3);
    }
  }

  // normalize and store a transposed: a_t[b][t][head*64 + c] (bf16)
  const int b = bh >> 3, head = bh & 7;
  bf16* aout = a_t + (size_t)b * LDIM * CDIM + head * CH;
#pragma unroll
  for (int r = 0; r < 8; r++) {
    float invl = 1.f / lrow[r];
    bf16* dst = aout + (size_t)(t0 + r + 8 * hh) * CDIM;
    dst[ 0 + m] = (bf16)(O0[r] * invl);
    dst[16 + m] = (bf16)(O1[r] * invl);
    dst[32 + m] = (bf16)(O2[r] * invl);
    dst[48 + m] = (bf16)(O3[r] * invl);
  }
}

// ---------------------------------------------------------------------------
// Kernel 4: out = x + proj_b + proj_w (512x512) @ a (512x4096), fp32 out.
// ---------------------------------------------------------------------------
__global__ __launch_bounds__(256) void proj_kernel(const bf16* __restrict__ wp_bf,
                                                   const float* __restrict__ proj_b,
                                                   const bf16* __restrict__ a_t,
                                                   const float* __restrict__ x,
                                                   float* __restrict__ out) {
  const int wave = threadIdx.x >> 5, lane = threadIdx.x & 31;
  const int m = lane & 15, hh = lane >> 4;
  const int ot = blockIdx.x * 8 + wave;   // 0..31
  const int lt = blockIdx.y;              // 0..255
  const int bb = blockIdx.z;
  const int o0 = ot * 16, l0 = lt * 16;

  const bf16* arow = wp_bf + (size_t)(o0 + m) * CDIM;
  const bf16* brow = a_t + ((size_t)bb * LDIM + l0 + m) * CDIM;

  v8f acc = {};
#pragma unroll 4
  for (int c0 = 0; c0 < CDIM; c0 += 32)
    acc = wmma_bf16(loadA(arow + c0, hh), loadB(brow + c0, hh), acc);

  const int l = l0 + m;
#pragma unroll
  for (int r = 0; r < 8; r++) {
    int o = o0 + r + 8 * hh;
    size_t idx = ((size_t)bb * CDIM + o) * LDIM + l;
    out[idx] = x[idx] + proj_b[o] + acc[r];
  }
}

// ---------------------------------------------------------------------------
extern "C" void kernel_launch(void* const* d_in, const int* in_sizes, int n_in,
                              void* d_out, int out_size, void* d_ws, size_t ws_size,
                              hipStream_t stream) {
  (void)in_sizes; (void)n_in; (void)out_size; (void)ws_size;
  const float* x      = (const float*)d_in[0];
  const float* norm_w = (const float*)d_in[1];
  const float* norm_b = (const float*)d_in[2];
  const float* qkv_w  = (const float*)d_in[3];
  const float* qkv_b  = (const float*)d_in[4];
  const float* proj_w = (const float*)d_in[5];
  const float* proj_b = (const float*)d_in[6];

  // Workspace carve (~42 MB, fits MI455X's 192 MB L2).
  char* p = (char*)d_ws;
  auto alloc = [&](size_t bytes) -> void* {
    void* r = (void*)p;
    p += (bytes + 255) & ~(size_t)255;
    return r;
  };
  bf16* wq_bf = (bf16*)alloc((size_t)OQKV * CDIM * 2);
  bf16* wp_bf = (bf16*)alloc((size_t)CDIM * CDIM * 2);
  bf16* h_t   = (bf16*)alloc((size_t)2 * LDIM * CDIM * 2);
  bf16* q_t   = (bf16*)alloc((size_t)16 * LDIM * CH * 2);
  bf16* k_t   = (bf16*)alloc((size_t)16 * LDIM * CH * 2);
  bf16* v_n   = (bf16*)alloc((size_t)16 * CH * LDIM * 2);
  bf16* a_t   = (bf16*)alloc((size_t)2 * LDIM * CDIM * 2);

  cvt_kernel<<<(OQKV * CDIM) / 256, 256, 0, stream>>>(qkv_w, proj_w, wq_bf, wp_bf);
  gn_kernel<<<64, 256, 0, stream>>>(x, norm_w, norm_b, h_t);
  qkv_kernel<<<dim3(12, 256, 2), 256, 0, stream>>>(wq_bf, qkv_b, h_t, q_t, k_t, v_n);
  attn_kernel<<<dim3(32, 16), 256, 0, stream>>>(q_t, k_t, v_n, a_t);
  proj_kernel<<<dim3(4, 256, 2), 256, 0, stream>>>(wp_bf, proj_b, a_t, x, (float*)d_out);
}
